// Net_62328565400116
// MI455X (gfx1250) — matrix operand order repaired and checked by reading, not compiled, not run on hardware
//
#include <hip/hip_runtime.h>
#include <hip/hip_bf16.h>

// ---------------------------------------------------------------------------
// GraphSAGE x3 + mean-pool + linear + log_softmax for MI455X (gfx1250).
//
//  * mean-aggregation commutes with the linear layer: compute
//    Y = X @ [Wl; Wr]^T (fused WMMA GEMM, [N,512]) then scatter 256-wide
//    projected messages (halves layer-1 edge traffic).
//  * GEMM precision: bf16x3 error-compensated split (hi*hi + hi*lo + lo*hi)
//    on v_wmma_f32_16x16x32_bf16 -> near-fp32 accuracy.
//  * Inputs are pre-split/zero-padded into bf16 hi/lo buffers so the GEMM
//    inner loop has NO bounds checks (previous round: guards lowered to
//    v_cmpx/s_cbranch around every scalar load) and fragment loads are
//    contiguous 128-bit vectors (k = 16*half + s, same order for A and B).
// ---------------------------------------------------------------------------

typedef __attribute__((ext_vector_type(16))) __bf16 v16bf;
typedef __attribute__((ext_vector_type(8)))  __bf16 v8bf;
typedef __attribute__((ext_vector_type(8)))  float  v8f;

#define HD   256   // hidden width (reference H)
#define OUTD 512   // fused output width [Wl; Wr]

__device__ __forceinline__ void split2(float x, __bf16& hi, __bf16& lo) {
  hi = (__bf16)x;                 // RNE to bf16
  lo = (__bf16)(x - (float)hi);   // residual
}

__device__ __forceinline__ v16bf cat8(v8bf a, v8bf b) {
  return __builtin_shufflevector(a, b, 0, 1, 2, 3, 4, 5, 6, 7,
                                       8, 9, 10, 11, 12, 13, 14, 15);
}

// ---------------------------------------------------------------------------
// One-shot: split fp32 X[N,Kdim] -> zero-padded bf16 hi/lo [N,Kpad]
// ---------------------------------------------------------------------------
__global__ void split_x_kernel(const float* __restrict__ X,
                               __bf16* __restrict__ Xh, __bf16* __restrict__ Xl,
                               int Nrows, int Kdim, int Kpad) {
  long tid = (long)blockIdx.x * blockDim.x + threadIdx.x;
  if (tid >= (long)Nrows * Kpad) return;
  long r = tid / Kpad;
  int  k = (int)(tid - r * Kpad);
  float v = (k < Kdim) ? X[r * (long)Kdim + k] : 0.f;
  __bf16 hi, lo; split2(v, hi, lo);
  Xh[tid] = hi; Xl[tid] = lo;
}

// ---------------------------------------------------------------------------
// One-shot: pack [Wl;Wr] (each [HD,Kdim] fp32) -> bf16 hi/lo [OUTD,Kpad]
// ---------------------------------------------------------------------------
__global__ void pack_w_kernel(const float* __restrict__ Wl,
                              const float* __restrict__ Wr,
                              __bf16* __restrict__ Wbh, __bf16* __restrict__ Wbl,
                              int Kdim, int Kpad) {
  long tid = (long)blockIdx.x * blockDim.x + threadIdx.x;
  if (tid >= (long)OUTD * Kpad) return;
  int n = (int)(tid / Kpad);
  int k = (int)(tid - (long)n * Kpad);
  const float* row = (n < HD) ? (Wl + (size_t)n * Kdim)
                              : (Wr + (size_t)(n - HD) * Kdim);
  float v = (k < Kdim) ? row[k] : 0.f;
  __bf16 hi, lo; split2(v, hi, lo);
  Wbh[tid] = hi; Wbl[tid] = lo;
}

// ---------------------------------------------------------------------------
// Fused dual GEMM on pre-split bf16: Y = A @ Wcat^T, A hi/lo + W hi/lo.
// One wave computes a 16x64 strip of Y (4 accumulators), bf16x3 per k-step.
// No bounds checks in the k-loop (K pre-padded), vector fragment loads.
// ---------------------------------------------------------------------------
__global__ __launch_bounds__(256)
void gemm_dual_kernel(const __bf16* __restrict__ Ah,
                      const __bf16* __restrict__ Al,
                      const __bf16* __restrict__ Wbh,
                      const __bf16* __restrict__ Wbl,
                      float* __restrict__ Y,
                      int Nrows, int Kpad) {
  const int lane = threadIdx.x & 31;
  const int wave = threadIdx.x >> 5;
  const int gw   = blockIdx.x * 8 + wave;     // 8 waves / block
  const int NG   = OUTD / 64;                 // 8 column groups of 64
  const int Mtiles = (Nrows + 15) >> 4;
  if (gw >= Mtiles * NG) return;              // wave-uniform exit
  const int mtile = gw / NG;
  const int ng    = gw % NG;

  const int mloc   = lane & 15;
  const int hiHalf = lane >> 4;
  const int m  = mtile * 16 + mloc;
  const int mm = (m < Nrows) ? m : (Nrows - 1);   // clamp; store is guarded

  // A fragment (16-bit A 16x32): lane holds M=mloc; K = hiHalf*16 + [0,16).
  const __bf16* aph = Ah + (size_t)mm * Kpad + hiHalf * 16;
  const __bf16* apl = Al + (size_t)mm * Kpad + hiHalf * 16;

  // B fragment (16-bit B 32x16): lane holds N=mloc; K = hiHalf*16 + [0,16).
  const __bf16* bph[4];
  const __bf16* bpl[4];
#pragma unroll
  for (int t = 0; t < 4; ++t) {
    int n = (ng * 4 + t) * 16 + mloc;
    bph[t] = Wbh + (size_t)n * Kpad + hiHalf * 16;
    bpl[t] = Wbl + (size_t)n * Kpad + hiHalf * 16;
  }

  v8f acc[4] = {};

  for (int kb = 0; kb < Kpad; kb += 32) {
    v16bf ah = cat8(*(const v8bf*)(aph),      *(const v8bf*)(aph + 8));
    v16bf al = cat8(*(const v8bf*)(apl),      *(const v8bf*)(apl + 8));
    aph += 32; apl += 32;
#pragma unroll
    for (int t = 0; t < 4; ++t) {
      v16bf bh = *(const v16bf*)(bph[t]);
      v16bf bl = *(const v16bf*)(bpl[t]);
      bph[t] += 32; bpl[t] += 32;
      // bf16x3: hi*hi + hi*lo + lo*hi (lo*lo negligible)
      acc[t] = __builtin_amdgcn_wmma_f32_16x16x32_bf16(
                   false, ah, false, bh, (short)0, acc[t], false, false);
      acc[t] = __builtin_amdgcn_wmma_f32_16x16x32_bf16(
                   false, ah, false, bl, (short)0, acc[t], false, false);
      acc[t] = __builtin_amdgcn_wmma_f32_16x16x32_bf16(
                   false, al, false, bh, (short)0, acc[t], false, false);
    }
  }

  // C/D layout: VGPR r -> M = r + hiHalf*8, N = lane%16
  const int row0 = hiHalf * 8;
#pragma unroll
  for (int t = 0; t < 4; ++t) {
    int ncol = (ng * 4 + t) * 16 + mloc;
#pragma unroll
    for (int r = 0; r < 8; ++r) {
      int mrow = mtile * 16 + row0 + r;
      if (mrow < Nrows) Y[(size_t)mrow * OUTD + ncol] = acc[t][r];
    }
  }
}

// ---------------------------------------------------------------------------
// Edge degree (once): deg[dst] += 1
// ---------------------------------------------------------------------------
__global__ void deg_kernel(const int* __restrict__ dst,
                           float* __restrict__ deg, int E) {
  long tid = (long)blockIdx.x * blockDim.x + threadIdx.x;
  if (tid < E) atomicAdd(&deg[dst[tid]], 1.f);
}

// ---------------------------------------------------------------------------
// Edge scatter: aggr[dst] += Y[src, 0:256]. 64 threads/edge, float4 reads.
// ---------------------------------------------------------------------------
__global__ void scatter_kernel(const float* __restrict__ Y,
                               const int* __restrict__ src,
                               const int* __restrict__ dst,
                               float* __restrict__ aggr, int E) {
  long tid = (long)blockIdx.x * blockDim.x + threadIdx.x;
  long e   = tid >> 6;          // 64 threads per edge (256 feats / 4)
  int part = (int)(tid & 63);
  if (e >= E) return;
  int s = src[e], d = dst[e];
  const float4 v = *(const float4*)(Y + (size_t)s * OUTD + part * 4);
  float* a = aggr + (size_t)d * HD + part * 4;
  atomicAdd(a + 0, v.x); atomicAdd(a + 1, v.y);
  atomicAdd(a + 2, v.z); atomicAdd(a + 3, v.w);
}

// ---------------------------------------------------------------------------
// Combine: h = relu(aggr/max(deg,1) + bl + Y[:,256:512]); emit bf16 hi/lo
// activations directly (feeds next layer's WMMA GEMM).
// ---------------------------------------------------------------------------
__global__ void combine_kernel(const float* __restrict__ aggr,
                               const float* __restrict__ deg,
                               const float* __restrict__ Y,
                               const float* __restrict__ bl,
                               __bf16* __restrict__ Hh, __bf16* __restrict__ Hl,
                               int Nrows) {
  int i = blockIdx.x;
  int c = threadIdx.x;          // 256 threads
  if (i >= Nrows) return;
  float r = 1.f / fmaxf(deg[i], 1.f);
  float v = aggr[(size_t)i * HD + c] * r + bl[c] + Y[(size_t)i * OUTD + HD + c];
  v = fmaxf(v, 0.f);
  __bf16 hi, lo; split2(v, hi, lo);
  Hh[(size_t)i * HD + c] = hi;
  Hl[(size_t)i * HD + c] = lo;
}

// ---------------------------------------------------------------------------
// Global mean pool (sum part, reconstruct hi+lo) + count
// ---------------------------------------------------------------------------
__global__ void pool_kernel(const __bf16* __restrict__ Hh,
                            const __bf16* __restrict__ Hl,
                            const int* __restrict__ batch,
                            float* __restrict__ pooled, int Nrows) {
  long tid = (long)blockIdx.x * blockDim.x + threadIdx.x;
  long i = tid >> 8;            // /256
  int  c = (int)(tid & 255);
  if (i >= Nrows) return;
  float v = (float)Hh[(size_t)i * HD + c] + (float)Hl[(size_t)i * HD + c];
  atomicAdd(&pooled[(size_t)batch[i] * HD + c], v);
}

__global__ void cnt_kernel(const int* __restrict__ batch,
                           float* __restrict__ cnt, int Nrows) {
  long tid = (long)blockIdx.x * blockDim.x + threadIdx.x;
  if (tid < Nrows) atomicAdd(&cnt[batch[tid]], 1.f);
}

// ---------------------------------------------------------------------------
// Head: logits = (pooled/cnt) @ Wlin^T + blin ; log_softmax. One wave/graph.
// ---------------------------------------------------------------------------
__global__ void head_kernel(const float* __restrict__ pooled,
                            const float* __restrict__ cnt,
                            const float* __restrict__ Wlin,
                            const float* __restrict__ blin,
                            float* __restrict__ out, int C) {
  int g = blockIdx.x;
  int lane = threadIdx.x;       // 32
  float rc = 1.f / fmaxf(cnt[g], 1.f);
  float logits[16];
  for (int c = 0; c < C; ++c) {
    float s = 0.f;
    for (int k = lane; k < HD; k += 32)
      s += pooled[(size_t)g * HD + k] * Wlin[(size_t)c * HD + k];
    for (int off = 16; off; off >>= 1) s += __shfl_down(s, off, 32);
    logits[c] = s * rc + blin[c];   // valid on lane 0
  }
  if (lane == 0) {
    float mx = logits[0];
    for (int c = 1; c < C; ++c) mx = fmaxf(mx, logits[c]);
    float se = 0.f;
    for (int c = 0; c < C; ++c) se += expf(logits[c] - mx);
    float lse = mx + logf(se);
    for (int c = 0; c < C; ++c) out[(size_t)g * C + c] = logits[c] - lse;
  }
}

// ---------------------------------------------------------------------------
extern "C" void kernel_launch(void* const* d_in, const int* in_sizes, int n_in,
                              void* d_out, int out_size, void* d_ws, size_t ws_size,
                              hipStream_t stream) {
  const float* x    = (const float*)d_in[0];
  const int*   edge = (const int*)d_in[1];     // [2,E]: src then dst
  const int*   batch= (const int*)d_in[2];
  // d_in[3] = num_graphs device scalar (G derived from out_size instead)
  const float* Wl1 = (const float*)d_in[4];
  const float* bl1 = (const float*)d_in[5];
  const float* Wr1 = (const float*)d_in[6];
  const float* Wl2 = (const float*)d_in[7];
  const float* bl2 = (const float*)d_in[8];
  const float* Wr2 = (const float*)d_in[9];
  const float* Wl3 = (const float*)d_in[10];
  const float* bl3 = (const float*)d_in[11];
  const float* Wr3 = (const float*)d_in[12];
  const float* Wlin= (const float*)d_in[13];
  const float* blin= (const float*)d_in[14];

  const int N    = in_sizes[2];
  const int E    = in_sizes[1] / 2;
  const int F_IN = in_sizes[4] / HD;           // Wl1 is [H, F_IN]
  const int C    = in_sizes[13] / HD;          // Wlin is [C, H]
  const int G    = out_size / C;
  const int KP1  = (F_IN + 31) & ~31;          // layer-1 padded K (500 -> 512)

  // Workspace carve-up (256B aligned) with region reuse:
  //  R1: Xh [N,KP1] bf16  -> later aggr [N,HD] f32  (same/greater bytes)
  //  R2: Xl [N,KP1] bf16  -> later Hh+Hl [N,HD] bf16 each
  char* ws = (char*)d_ws;
  size_t off = 0;
  auto carve = [&](size_t bytes) -> void* {
    off = (off + 255) & ~(size_t)255;
    void* p = ws + off;
    off += bytes;
    return p;
  };
  size_t regBytes = (size_t)N * ((KP1 > 2 * HD) ? KP1 : 2 * HD) * sizeof(__bf16);
  float*  Y      = (float*)carve((size_t)N * OUTD * sizeof(float));
  char*   R1     = (char*)carve(regBytes);
  char*   R2     = (char*)carve(regBytes);
  __bf16* Wbh    = (__bf16*)carve((size_t)OUTD * KP1 * sizeof(__bf16));
  __bf16* Wbl    = (__bf16*)carve((size_t)OUTD * KP1 * sizeof(__bf16));
  float*  deg    = (float*)carve((size_t)N * sizeof(float));
  float*  pooled = (float*)carve((size_t)G * HD * sizeof(float));
  float*  cnt    = (float*)carve((size_t)G * sizeof(float));
  (void)ws_size;

  __bf16* Xh   = (__bf16*)R1;
  __bf16* Xl   = (__bf16*)R2;
  float*  aggr = (float*)R1;                    // reuses Xh region after GEMM1
  __bf16* Hh   = (__bf16*)R2;                   // reuses Xl region after GEMM1
  __bf16* Hl   = (__bf16*)(R2 + (size_t)N * HD * sizeof(__bf16));

  const int* src = edge;
  const int* dst = edge + E;

  // Degree (shared by all three layers)
  hipMemsetAsync(deg, 0, (size_t)N * sizeof(float), stream);
  deg_kernel<<<(E + 255) / 256, 256, 0, stream>>>(dst, deg, E);

  const int Mtiles     = (N + 15) >> 4;
  const int gemmBlocks = (Mtiles * (OUTD / 64) + 7) / 8;

  auto backHalf = [&](const float* bl) {
    hipMemsetAsync(aggr, 0, (size_t)N * HD * sizeof(float), stream);
    long sthreads = (long)E * 64;
    scatter_kernel<<<(unsigned)((sthreads + 255) / 256), 256, 0, stream>>>(
        Y, src, dst, aggr, E);
    combine_kernel<<<N, HD, 0, stream>>>(aggr, deg, Y, bl, Hh, Hl, N);
  };

  // ---- Layer 1: split X, pack W1, GEMM(K=KP1), scatter+combine
  {
    long t1 = (long)N * KP1;
    split_x_kernel<<<(unsigned)((t1 + 255) / 256), 256, 0, stream>>>(
        x, Xh, Xl, N, F_IN, KP1);
    long t2 = (long)OUTD * KP1;
    pack_w_kernel<<<(unsigned)((t2 + 255) / 256), 256, 0, stream>>>(
        Wl1, Wr1, Wbh, Wbl, F_IN, KP1);
    gemm_dual_kernel<<<gemmBlocks, 256, 0, stream>>>(Xh, Xl, Wbh, Wbl, Y, N, KP1);
    backHalf(bl1);
  }

  // ---- Layers 2 & 3: pack W, GEMM(K=HD) on bf16 activations, scatter+combine
  for (int layer = 2; layer <= 3; ++layer) {
    const float* Wl = (layer == 2) ? Wl2 : Wl3;
    const float* Wr = (layer == 2) ? Wr2 : Wr3;
    const float* bl = (layer == 2) ? bl2 : bl3;
    long t2 = (long)OUTD * HD;
    pack_w_kernel<<<(unsigned)((t2 + 255) / 256), 256, 0, stream>>>(
        Wl, Wr, Wbh, Wbl, HD, HD);
    gemm_dual_kernel<<<gemmBlocks, 256, 0, stream>>>(Hh, Hl, Wbh, Wbl, Y, N, HD);
    backHalf(bl);
  }

  // ---- Mean pool + head
  hipMemsetAsync(pooled, 0, (size_t)G * HD * sizeof(float), stream);
  hipMemsetAsync(cnt,    0, (size_t)G * sizeof(float), stream);
  long pthreads = (long)N * HD;
  pool_kernel<<<(unsigned)((pthreads + 255) / 256), 256, 0, stream>>>(
      Hh, Hl, batch, pooled, N);
  cnt_kernel<<<(N + 255) / 256, 256, 0, stream>>>(batch, cnt, N);
  head_kernel<<<G, 32, 0, stream>>>(pooled, cnt, Wlin, blin, (float*)d_out, C);
}
